// EmbeddingSoftAttention_13357348291341
// MI455X (gfx1250) — matrix operand order repaired and checked by reading, not compiled, not run on hardware
//
#include <hip/hip_runtime.h>
#include <hip/hip_bf16.h>

#define BATCH  256
#define LMAX_  512
#define NCOND_ 300
#define NHID_  600
#define KPAD   320          // 10 * 32 (K padded with zeros)
#define KSTEPS 10
#define AROW   328          // LDS row stride in halves; 656B = 41*16 (16B aligned, bank-spread)
#define NVEC   75           // float4s per context row (300/4)
#define TILE4  (16 * NVEC)  // 1200 float4 loads per 16-token tile
#define NLD    5            // ceil(1200/256) staged float4s per thread

typedef __attribute__((ext_vector_type(16))) _Float16 v16h;
typedef __attribute__((ext_vector_type(4)))  _Float16 v4h;
typedef __attribute__((ext_vector_type(8)))  float    v8f;

union AFrag { v16h v; uint4 q[2]; };

// ---------------------------------------------------------------------------
// Phase 1: pooled[b][n] = (1/len) * sum_{l<len} relu(ctx[b][l][:] @ W1[:,n] + b1[n])
// 8 waves/block = 2 waves/SIMD -> tell the scheduler (launch_bounds min-waves=2)
// so it schedules for latency, not register pressure.
// ---------------------------------------------------------------------------
__global__ void __launch_bounds__(256, 2)
esa_pool_gemm(const float* __restrict__ ctx, const int* __restrict__ lengths,
              const float* __restrict__ W1, const float* __restrict__ b1,
              float* __restrict__ pooled)
{
    __shared__ __align__(16) _Float16 As[2][16 * AROW];   // double-buffered A tile

    const int b    = blockIdx.y;
    const int tid  = threadIdx.x;
    const int lane = tid & 31;
    const int wave = tid >> 5;
    const int nbase = blockIdx.x * 128 + wave * 16;
    const int ncol  = nbase + (lane & 15);
    const int hi    = lane >> 4;
    const int mrowL = lane & 15;

    int len = lengths[b];
    if (len < 1) len = 1;
    const int msteps = (len + 15) >> 4;

    const float* ctxb = ctx + (size_t)b * LMAX_ * NCOND_;

    // ---- per-thread tile-copy addressing, hoisted out of the M-loop --------
    int  offv[NLD];     // float offset of this thread's j-th float4 within a tile
    int  ldsoff[NLD];   // half-element offset within an LDS tile buffer
    bool act[NLD];
    #pragma unroll
    for (int j = 0; j < NLD; ++j) {
        const int i = tid + j * 256;
        act[j] = (i < TILE4);
        const int m = i / NVEC;
        const int q = i - m * NVEC;
        offv[j]   = m * NCOND_ + 4 * q;
        ldsoff[j] = m * AROW   + 4 * q;
    }

    // ---- B fragments (W1 tile) resident in registers for the whole kernel.
    // 16-bit B 32x16 layout: lane = column n; elem e<8 <-> k=K0+hi*8+e, e>=8 <-> k=K0+16+hi*8+(e-8)
    const bool ncol_ok = (ncol < NHID_);
    v16h breg[KSTEPS];
    #pragma unroll
    for (int t = 0; t < KSTEPS; ++t) {
        v16h bb;
        const int kb = t * 32 + hi * 8;
        #pragma unroll
        for (int e = 0; e < 16; ++e) {
            const int k = kb + e + (e >= 8 ? 8 : 0);
            float f = (ncol_ok && k < NCOND_) ? W1[(size_t)k * NHID_ + ncol] : 0.0f;
            bb[e] = (_Float16)f;
        }
        breg[t] = bb;
    }
    const float b1v = ncol_ok ? b1[ncol] : 0.0f;

    // ---- zero the read K-pad region [NCOND_, KPAD) of both buffers (once)
    for (int i = tid; i < 2 * 16 * (KPAD - NCOND_); i += 256) {
        const int buf = i / (16 * (KPAD - NCOND_));
        const int r   = i - buf * 16 * (KPAD - NCOND_);
        const int m   = r / (KPAD - NCOND_);
        const int k   = NCOND_ + (r - m * (KPAD - NCOND_));
        As[buf][m * AROW + k] = (_Float16)0.0f;
    }

    // ---- prologue: stage + store tile 0
    float4 stage[NLD];
    #pragma unroll
    for (int j = 0; j < NLD; ++j)
        if (act[j]) stage[j] = *(const float4*)(ctxb + offv[j]);
    #pragma unroll
    for (int j = 0; j < NLD; ++j) {
        if (act[j]) {
            v4h h;
            h[0] = (_Float16)stage[j].x;  h[1] = (_Float16)stage[j].y;
            h[2] = (_Float16)stage[j].z;  h[3] = (_Float16)stage[j].w;
            *(v4h*)(&As[0][ldsoff[j]]) = h;
        }
    }
    __syncthreads();

    v8f colsum8 = {0.f, 0.f, 0.f, 0.f, 0.f, 0.f, 0.f, 0.f};

    for (int ms = 0; ms < msteps; ++ms) {
        const int cur  = ms & 1;
        const bool more = (ms + 1) < msteps;

        // stage next tile's global loads (consumed after compute)
        if (more) {
            const float* tb = ctxb + (size_t)(ms + 1) * 16 * NCOND_;
            #pragma unroll
            for (int j = 0; j < NLD; ++j)
                if (act[j]) stage[j] = *(const float4*)(tb + offv[j]);
        }
        // prefetch one tile deeper (global_prefetch_b8) to warm L2/L0
        if (ms + 2 < msteps) {
            const float* pf = ctxb + (size_t)(ms + 2) * 16 * NCOND_;
            #pragma unroll
            for (int j = 0; j < NLD; ++j)
                if (act[j]) __builtin_prefetch(pf + offv[j], 0, 3);
        }

        // ---- load ALL 10 A-fragments, then 10 back-to-back WMMAs.
        // A 16-bit 16x32 layout: 8 halves at K0+hi*8 and 8 halves at K0+16+hi*8.
        const _Float16* arow = &As[cur][mrowL * AROW + hi * 8];
        AFrag a[KSTEPS];
        #pragma unroll
        for (int t = 0; t < KSTEPS; ++t) {
            a[t].q[0] = *(const uint4*)(arow + t * 32);
            a[t].q[1] = *(const uint4*)(arow + t * 32 + 16);
        }
        v8f c = {b1v, b1v, b1v, b1v, b1v, b1v, b1v, b1v};  // bias folded into C init
        #pragma unroll
        for (int t = 0; t < KSTEPS; ++t) {
            c = __builtin_amdgcn_wmma_f32_16x16x32_f16(
                    false, a[t].v, false, breg[t], (short)0, c, false, false);
        }

        // relu + ragged-row mask into 8 independent accumulators
        const int rem = len - (ms * 16 + hi * 8);   // valid rows in this half
        #pragma unroll
        for (int r = 0; r < 8; ++r) {
            float h = c[r];
            h = h > 0.0f ? h : 0.0f;
            colsum8[r] += (r < rem) ? h : 0.0f;
        }

        // drain next tile into the alternate buffer, then sync
        if (more) {
            _Float16* dst = As[cur ^ 1];
            #pragma unroll
            for (int j = 0; j < NLD; ++j) {
                if (act[j]) {
                    v4h h;
                    h[0] = (_Float16)stage[j].x;  h[1] = (_Float16)stage[j].y;
                    h[2] = (_Float16)stage[j].z;  h[3] = (_Float16)stage[j].w;
                    *(v4h*)(dst + ldsoff[j]) = h;
                }
            }
        }
        __syncthreads();
    }

    float colsum = ((colsum8[0] + colsum8[1]) + (colsum8[2] + colsum8[3]))
                 + ((colsum8[4] + colsum8[5]) + (colsum8[6] + colsum8[7]));
    const float other = __shfl_xor(colsum, 16, 32);
    const float total = colsum + other;
    if (hi == 0 && ncol_ok) {
        pooled[(size_t)b * NHID_ + ncol] = total / (float)len;
    }
}

// ---------------------------------------------------------------------------
// Phase 2: out[b][c] = sigmoid(pooled[b][:] @ Wa[:,c] + ba[c]) * x[b][c]
// ---------------------------------------------------------------------------
__global__ void __launch_bounds__(256)
esa_gate(const float* __restrict__ pooled, const float* __restrict__ Wa,
         const float* __restrict__ ba, const float* __restrict__ x,
         float* __restrict__ out)
{
    __shared__ float p[NHID_];
    const int b   = blockIdx.x;
    const int tid = threadIdx.x;

    for (int i = tid; i < NHID_; i += 256) p[i] = pooled[(size_t)b * NHID_ + i];
    __syncthreads();

    for (int c = tid; c < NCOND_; c += 256) {
        float acc = ba[c];
        #pragma unroll 4
        for (int k = 0; k < NHID_; ++k) acc += p[k] * Wa[(size_t)k * NCOND_ + c];
        const float g = 1.0f / (1.0f + __expf(-acc));
        out[(size_t)b * NCOND_ + c] = g * x[(size_t)b * NCOND_ + c];
    }
}

// ---------------------------------------------------------------------------
extern "C" void kernel_launch(void* const* d_in, const int* in_sizes, int n_in,
                              void* d_out, int out_size, void* d_ws, size_t ws_size,
                              hipStream_t stream)
{
    (void)in_sizes; (void)n_in; (void)out_size; (void)ws_size;

    const float* x       = (const float*)d_in[0];   // [256, 300]
    const float* ctx     = (const float*)d_in[1];   // [256, 512, 300]
    const int*   lengths = (const int*)  d_in[2];   // [256]
    const float* W1      = (const float*)d_in[3];   // [300, 600]
    const float* b1      = (const float*)d_in[4];   // [600]
    const float* Wa      = (const float*)d_in[5];   // [600, 300]
    const float* ba      = (const float*)d_in[6];   // [300]
    float*       out     = (float*)d_out;           // [256, 300]
    float*       pooled  = (float*)d_ws;            // [256, 600] scratch (614 KB)

    dim3 grid1(5, BATCH);
    esa_pool_gemm<<<grid1, 256, 0, stream>>>(ctx, lengths, W1, b1, pooled);
    esa_gate<<<BATCH, 256, 0, stream>>>(pooled, Wa, ba, x, out);
}
